// SwitchFFN_78219944395075
// MI455X (gfx1250) — matrix operand order repaired and checked by reading, not compiled, not run on hardware
//
#include <hip/hip_runtime.h>
#include <hip/hip_bf16.h>
#include <stdint.h>

// ---- problem constants (match reference) ----
constexpr int kD = 1024, kF = 4096, kE = 8;
constexpr int kT = 4 * 2048;                       // 8192 tokens
constexpr int kCap = (int)(kT / kE * 1.25);        // 1280
constexpr int kNB = kT / 256;                      // 32 histogram blocks
constexpr float kAuxAlpha = 0.01f;

typedef __attribute__((ext_vector_type(16))) __bf16 bf16x16;
typedef __attribute__((ext_vector_type(8)))  float  f32x8;

__device__ __forceinline__ unsigned short f32_to_bf16(float f) {
  unsigned u = __float_as_uint(f);
  u += 0x7fffu + ((u >> 16) & 1u);   // round-to-nearest-even
  return (unsigned short)(u >> 16);
}

__device__ __forceinline__ unsigned pack_bf16x2(float lo, float hi) {
  return (unsigned)f32_to_bf16(lo) | ((unsigned)f32_to_bf16(hi) << 16);
}

// ---------------- router: wave32-per-token ----------------
__global__ __launch_bounds__(256) void k_router(
    const float* __restrict__ x, const float* __restrict__ rw,
    const float* __restrict__ rb, int* __restrict__ eidx,
    float* __restrict__ gate, float* __restrict__ ptok) {
  const int tid = threadIdx.x;
  const int wave = tid >> 5, lane = tid & 31;
  const int t = blockIdx.x * 8 + wave;
  float acc[kE];
#pragma unroll
  for (int e = 0; e < kE; ++e) acc[e] = 0.f;
  const float* xr = x + (size_t)t * kD;
  for (int d = lane; d < kD; d += 32) {
    float xv = xr[d];
#pragma unroll
    for (int e = 0; e < kE; ++e) acc[e] += xv * rw[d * kE + e];
  }
#pragma unroll
  for (int off = 16; off >= 1; off >>= 1)
#pragma unroll
    for (int e = 0; e < kE; ++e) acc[e] += __shfl_xor(acc[e], off, 32);
  if (lane == 0) {
    float mx = -1e30f; int am = 0;
#pragma unroll
    for (int e = 0; e < kE; ++e) {
      float l = acc[e] + rb[e];
      acc[e] = l;
      if (l > mx) { mx = l; am = e; }   // strict > keeps first max (jnp.argmax)
    }
    float s = 0.f;
#pragma unroll
    for (int e = 0; e < kE; ++e) { acc[e] = __expf(acc[e] - mx); s += acc[e]; }
    float inv = 1.f / s;
#pragma unroll
    for (int e = 0; e < kE; ++e) ptok[(size_t)t * kE + e] = acc[e] * inv;
    eidx[t] = am;
    gate[t] = acc[am] * inv;
  }
}

// ---------------- histogram / scan ----------------
__global__ __launch_bounds__(256) void k_hist(const int* __restrict__ eidx,
                                              int* __restrict__ bhist) {
  __shared__ int h[kE];
  const int tid = threadIdx.x;
  if (tid < kE) h[tid] = 0;
  __syncthreads();
  atomicAdd(&h[eidx[blockIdx.x * 256 + tid]], 1);
  __syncthreads();
  if (tid < kE) bhist[blockIdx.x * kE + tid] = h[tid];
}

__global__ void k_scan(const int* __restrict__ bhist, const float* __restrict__ ptok,
                       int* __restrict__ boff, int* __restrict__ counts,
                       float* __restrict__ psum) {
  const int e = threadIdx.x;
  if (e >= kE) return;
  int run = 0;
  for (int b = 0; b < kNB; ++b) { boff[b * kE + e] = run; run += bhist[b * kE + e]; }
  counts[e] = run;
  float s = 0.f;
  for (int t = 0; t < kT; ++t) s += ptok[(size_t)t * kE + e];
  psum[e] = s;
}

// ---------------- position + scatter (f32 -> bf16, packed stores) ----------------
__global__ __launch_bounds__(256) void k_scatter(
    const float* __restrict__ x, const int* __restrict__ eidx,
    const int* __restrict__ boff, int* __restrict__ pos,
    unsigned short* __restrict__ xin) {
  __shared__ int sIdx[256];
  __shared__ int sPos[256];
  const int tid = threadIdx.x;
  const int t = blockIdx.x * 256 + tid;
  const int my = eidx[t];
  sIdx[tid] = my;
  __syncthreads();
  int r = 0;
  for (int j = 0; j < tid; ++j) r += (sIdx[j] == my);
  const int p = boff[blockIdx.x * kE + my] + r;
  pos[t] = p;
  sPos[tid] = p;
  __syncthreads();
  for (int i = 0; i < 256; ++i) {
    const int e = sIdx[i];
    const int p2 = sPos[i];
    if (p2 < kCap) {
      const float* src = x + (size_t)(blockIdx.x * 256 + i) * kD;
      unsigned* dst = (unsigned*)(xin + ((size_t)e * kCap + p2) * kD);
#pragma unroll
      for (int d = tid; d < kD / 2; d += 256)
        dst[d] = pack_bf16x2(src[2 * d], src[2 * d + 1]);
    }
  }
}

// ---------------- bf16 WMMA GEMM: C[e] = act(A[e] @ W[e] + bias[e]) ----------------
// Block tile 128x128, K-step 32, double-buffered LDS; 8 waves in 4(M) x 2(N);
// wave tile 32x64 = 2x4 WMMA per K-step. A tile staged with async global->LDS DMA.
template <int K, int N, bool RELU_BF16>
__global__ __launch_bounds__(256) void k_gemm(
    const unsigned short* __restrict__ Aall,  // [E][kCap][K] bf16
    const float* __restrict__ Wall,           // [E][K][N]    f32
    const float* __restrict__ Ball,           // [E][N]       f32
    void* __restrict__ Call) {                // [E][kCap][N] bf16 or f32
  constexpr int LDSS = 48;                    // padded row stride (16B aligned)
  __shared__ unsigned short As[2][128 * LDSS];
  __shared__ unsigned short Bs[2][128 * LDSS];   // stored transposed: Bs[n][k]

  const int e = blockIdx.z;
  const int n0 = blockIdx.x * 128;
  const int m0 = blockIdx.y * 128;
  const unsigned short* A = Aall + (size_t)e * kCap * K;
  const float* W = Wall + (size_t)e * K * N;
  const float* bias = Ball + (size_t)e * N;

  const int tid = threadIdx.x;
  const int lane = tid & 31;
  const int wv = tid >> 5;
  const int wm = wv & 3;         // 0..3 along M
  const int wn = wv >> 2;        // 0..1 along N
  const int lh = lane & 15;
  const int lq = lane >> 4;      // half-wave select

  f32x8 acc[2][4] = {};

  const int arow = tid >> 2;           // 0..63
  const int achunk = (tid & 3) * 8;    // 8 bf16 = 16B chunks
  const int bn = tid & 127;            // staged column
  const int bk2 = tid >> 7;            // base k-pair (0..1)

  // stage A tile: async DMA global -> LDS (bf16 copy, no VGPR round trip)
  auto stageA = [&](int buf, int kk) {
#pragma unroll
    for (int r2 = 0; r2 < 2; ++r2) {
      const int r = arow + r2 * 64;
      const unsigned short* gp = A + (size_t)(m0 + r) * K + kk + achunk;
      const unsigned lds = (unsigned)(uintptr_t)&As[buf][r * LDSS + achunk];
      asm volatile("global_load_async_to_lds_b128 %0, %1, off"
                   :: "v"(lds), "v"(gp) : "memory");
    }
  };
  // stage W tile: coalesced f32 loads along N, convert+pack, transposed b32 stores
  auto stageB = [&](int buf, int kk) {
#pragma unroll
    for (int i = 0; i < 8; ++i) {
      const int k2 = bk2 + i * 2;      // 0..15 (pairs of K)
      const float w0 = W[(size_t)(kk + 2 * k2) * N + n0 + bn];
      const float w1 = W[(size_t)(kk + 2 * k2 + 1) * N + n0 + bn];
      *(unsigned*)&Bs[buf][bn * LDSS + 2 * k2] = pack_bf16x2(w0, w1);
    }
  };
  auto compute = [&](int buf) {
    bf16x16 bfr[4];
#pragma unroll
    for (int tn = 0; tn < 4; ++tn) {
      const unsigned short* bp = &Bs[buf][(wn * 64 + tn * 16 + lh) * LDSS + lq * 8];
      ((uint4*)&bfr[tn])[0] = *(const uint4*)(bp);
      ((uint4*)&bfr[tn])[1] = *(const uint4*)(bp + 16);
    }
#pragma unroll
    for (int tm = 0; tm < 2; ++tm) {
      // A 16x32 bf16 layout: lanes 0-15 hold K {0..7,16..23}, lanes 16-31 {8..15,24..31}
      const unsigned short* ap = &As[buf][(wm * 32 + tm * 16 + lh) * LDSS + lq * 8];
      bf16x16 afr;
      ((uint4*)&afr)[0] = *(const uint4*)(ap);
      ((uint4*)&afr)[1] = *(const uint4*)(ap + 16);
#pragma unroll
      for (int tn = 0; tn < 4; ++tn)
        acc[tm][tn] = __builtin_amdgcn_wmma_f32_16x16x32_bf16(
            false, afr, false, bfr[tn], (short)0, acc[tm][tn], false, false);
    }
  };

  stageA(0, 0);
  stageB(0, 0);
  asm volatile("s_wait_asynccnt 0x0" ::: "memory");
  __syncthreads();

  int cur = 0;
  for (int kk = 0; kk < K; kk += 32) {
    const int nxt = cur ^ 1;
    if (kk + 32 < K) {            // prefetch next slab into the other buffer
      stageA(nxt, kk + 32);
      stageB(nxt, kk + 32);
    }
    compute(cur);
    asm volatile("s_wait_asynccnt 0x0" ::: "memory");
    __syncthreads();
    cur = nxt;
  }

  // epilogue: C/D layout — VGPR vi, lanes 0-15: M=vi, lanes 16-31: M=vi+8; N=lane&15
#pragma unroll
  for (int tn = 0; tn < 4; ++tn) {
    const int nc = n0 + wn * 64 + tn * 16 + lh;
    const float b = bias[nc];
#pragma unroll
    for (int tm = 0; tm < 2; ++tm) {
      const int mrBase = m0 + wm * 32 + tm * 16 + lq * 8;
#pragma unroll
      for (int vi = 0; vi < 8; ++vi) {
        float v = acc[tm][tn][vi] + b;
        const size_t co = ((size_t)e * kCap + (mrBase + vi)) * N + nc;
        if (RELU_BF16)
          ((unsigned short*)Call)[co] = f32_to_bf16(fmaxf(v, 0.f));
        else
          ((float*)Call)[co] = v;
      }
    }
  }
}

// ---------------- gather + gate ----------------
__global__ __launch_bounds__(256) void k_gather(
    const float* __restrict__ eout, const int* __restrict__ eidx,
    const int* __restrict__ pos, const float* __restrict__ gate,
    float* __restrict__ out) {
  const int t = blockIdx.x;
  const int tid = threadIdx.x;
  const int e = eidx[t];
  const int p = pos[t];
  const float g = gate[t];
  float* dst = out + (size_t)t * kD;
  if (p < kCap) {
    const float* src = eout + ((size_t)e * kCap + p) * kD;
    for (int d = tid; d < kD; d += 256) dst[d] = g * src[d];
  } else {
    for (int d = tid; d < kD; d += 256) dst[d] = 0.f;
  }
}

// ---------------- scalars: aux_loss, drop_ratio, counts ----------------
__global__ void k_finalize(const int* __restrict__ counts,
                           const float* __restrict__ psum,
                           float* __restrict__ tail) {
  if (threadIdx.x != 0 || blockIdx.x != 0) return;
  float dotv = 0.f;
  int dropped = 0;
  for (int e = 0; e < kE; ++e) {
    dotv += ((float)counts[e] / (float)kT) * (psum[e] / (float)kT);
    const int ov = counts[e] - kCap;
    dropped += ov > 0 ? ov : 0;
  }
  tail[0] = kAuxAlpha * (float)kE * dotv;
  tail[1] = (float)dropped / (float)kT;
  for (int e = 0; e < kE; ++e) tail[2 + e] = (float)counts[e];
}

extern "C" void kernel_launch(void* const* d_in, const int* in_sizes, int n_in,
                              void* d_out, int out_size, void* d_ws, size_t ws_size,
                              hipStream_t stream) {
  const float* x        = (const float*)d_in[0];
  const float* router_w = (const float*)d_in[1];
  const float* router_b = (const float*)d_in[2];
  const float* w_in     = (const float*)d_in[3];
  const float* b_in     = (const float*)d_in[4];
  const float* w_out    = (const float*)d_in[5];
  const float* b_out    = (const float*)d_in[6];
  float* out = (float*)d_out;

  char* ws = (char*)d_ws;
  size_t off = 0;
  auto take = [&](size_t bytes) -> void* {
    void* p = (void*)(ws + off);
    off += (bytes + 255) & ~(size_t)255;
    return p;
  };
  int*   eidx  = (int*)take((size_t)kT * 4);
  float* gate  = (float*)take((size_t)kT * 4);
  int*   pos   = (int*)take((size_t)kT * 4);
  float* ptok  = (float*)take((size_t)kT * kE * 4);
  int*   bhist = (int*)take((size_t)kNB * kE * 4);
  int*   boff  = (int*)take((size_t)kNB * kE * 4);
  int*   counts= (int*)take((size_t)kE * 4);
  float* psum  = (float*)take((size_t)kE * 4);
  unsigned short* xin  = (unsigned short*)take((size_t)kE * kCap * kD * 2);
  unsigned short* hbuf = (unsigned short*)take((size_t)kE * kCap * kF * 2);
  float*          eout = (float*)take((size_t)kE * kCap * kD * 4);

  k_router <<<kT / 8, 256, 0, stream>>>(x, router_w, router_b, eidx, gate, ptok);
  k_hist   <<<kNB,    256, 0, stream>>>(eidx, bhist);
  k_scan   <<<1,       32, 0, stream>>>(bhist, ptok, boff, counts, psum);
  k_scatter<<<kNB,    256, 0, stream>>>(x, eidx, boff, pos, xin);
  k_gemm<kD, kF, true >
      <<<dim3(kF / 128, kCap / 128, kE), 256, 0, stream>>>(xin, w_in, b_in, (void*)hbuf);
  k_gemm<kF, kD, false>
      <<<dim3(kD / 128, kCap / 128, kE), 256, 0, stream>>>(hbuf, w_out, b_out, (void*)eout);
  k_gather <<<kT,     256, 0, stream>>>(eout, eidx, pos, gate, out);
  k_finalize<<<1,       1, 0, stream>>>(counts, psum, out + (size_t)kT * kD);
}
